// GIBGCN_41618233098322
// MI455X (gfx1250) — matrix operand order repaired and checked by reading, not compiled, not run on hardware
//
#include <hip/hip_runtime.h>

#define B_   64
#define N_   2048
#define P3_  128
#define HID_ 128
#define F_   256      // P3 + HID
#define E_   32768

typedef __attribute__((ext_vector_type(16))) _Float16 v16h;
typedef __attribute__((ext_vector_type(8)))  float    v8f;

// d_out layout (floats), tuple order: out, pos_embedding, graph_embedding, pos_penalty, assignment
#define OUT_OFF 0
#define POS_OFF 8192
#define GE_OFF  16384
#define PEN_OFF 24576
#define ASN_OFF 24577   // NOTE: odd offset -> assignment stores must be scalar floats

// d_ws layout (floats)
#define WS_POS  0            // [64*16*128] partial pos sums
#define WS_GE   131072       // [64*16*128] partial graph sums
#define WS_PEN  262144       // [64] per-graph penalty

// -------------------------------------------------------------------------
// Kernel 1: assignment = softmax(W_c2 · relu(W_c1 · emb + b_c1) + b_c2)
// 1024 blocks x 256 threads (8 waves). Each wave: 16 rows x 128 cols via
// 8 x v_wmma_f32_16x16x32_f16 tiles, K=256 in 8 steps of 32.
// -------------------------------------------------------------------------
__global__ void __launch_bounds__(256)
k_assign(const float* __restrict__ emb,
         const float* __restrict__ Wc1, const float* __restrict__ bc1,
         const float* __restrict__ Wc2, const float* __restrict__ bc2,
         float* __restrict__ out)
{
    __shared__ _Float16 sWk[128 * 40];   // 32-wide K slice of W_c1 (f16), pitch 40 halves
    __shared__ float sW2[2 * 128];
    __shared__ float sBc1[128];
    __shared__ float sB2[2];

    const int tid   = threadIdx.x;
    const int wave  = tid >> 5;
    const int lane  = tid & 31;
    const int laneN = lane & 15;
    const int half  = lane >> 4;

    if (tid < 128) sBc1[tid] = bc1[tid];
    sW2[tid] = Wc2[tid];                 // 256 threads == 2*128 weights
    if (tid < 2) sB2[tid] = bc2[tid];

    const int m0 = blockIdx.x * 128 + wave * 16;   // first row of this wave's tile

    v8f acc[8] = {};

    for (int kb = 0; kb < 8; ++kb) {
        __syncthreads();
        // stage W_c1[:, kb*32 .. kb*32+32) into LDS as f16 (2 threads per p-row)
        {
            const int p      = tid >> 1;
            const int kkbase = (tid & 1) * 16;
            const float* wsrc = Wc1 + (size_t)p * F_ + kb * 32 + kkbase;
            _Float16* wdst = &sWk[p * 40 + kkbase];
            #pragma unroll
            for (int j = 0; j < 16; ++j) wdst[j] = (_Float16)wsrc[j];
        }
        __syncthreads();

        // A tile: row = laneN, K contiguous: k = 16*half + j
        const float* arow = emb + (size_t)(m0 + laneN) * F_ + kb * 32 + half * 16;
        v16h a;
        #pragma unroll
        for (int j = 0; j < 8; ++j) {
            a[j]     = (_Float16)arow[j];
            a[8 + j] = (_Float16)arow[8 + j];
        }

        #pragma unroll
        for (int t = 0; t < 8; ++t) {
            const _Float16* bp = &sWk[(t * 16 + laneN) * 40 + half * 16];
            v16h bv;
            #pragma unroll
            for (int j = 0; j < 16; ++j) bv[j] = bp[j];
            acc[t] = __builtin_amdgcn_wmma_f32_16x16x32_f16(
                false, a, false, bv, (short)0, acc[t], false, false);
        }
    }

    // bias + relu, in accumulator layout: element r of acc[t] is row (8*half+r), col (16t+laneN)
    #pragma unroll
    for (int t = 0; t < 8; ++t) {
        const float bc = sBc1[t * 16 + laneN];
        #pragma unroll
        for (int r = 0; r < 8; ++r) acc[t][r] = fmaxf(acc[t][r] + bc, 0.f);
    }

    float w0[8], w1[8];
    #pragma unroll
    for (int t = 0; t < 8; ++t) {
        w0[t] = sW2[t * 16 + laneN];
        w1[t] = sW2[128 + t * 16 + laneN];
    }

    #pragma unroll
    for (int r = 0; r < 8; ++r) {
        float l0 = 0.f, l1 = 0.f;
        #pragma unroll
        for (int t = 0; t < 8; ++t) {
            l0 += acc[t][r] * w0[t];
            l1 += acc[t][r] * w1[t];
        }
        // butterfly reduce over the 16 lanes of this half-wave
        #pragma unroll
        for (int off = 8; off >= 1; off >>= 1) {
            l0 += __shfl_xor(l0, off, 32);
            l1 += __shfl_xor(l1, off, 32);
        }
        if (laneN == r) {
            l0 += sB2[0]; l1 += sB2[1];
            const float mx = fmaxf(l0, l1);
            const float e0 = expf(l0 - mx), e1 = expf(l1 - mx);
            const float inv = 1.f / (e0 + e1);
            const size_t m = (size_t)(m0 + half * 8 + r);
            out[ASN_OFF + m * 2 + 0] = e0 * inv;
            out[ASN_OFF + m * 2 + 1] = e1 * inv;
        }
    }
}

// -------------------------------------------------------------------------
// Kernel 2a: partial sums over n-chunks for pos_embedding / graph_embedding
// grid 1024 (= 64 b * 16 chunks) x 128 threads (p)
// -------------------------------------------------------------------------
__global__ void __launch_bounds__(128)
k_embed_part(const float* __restrict__ x, const float* __restrict__ dout,
             float* __restrict__ ws)
{
    const int b = blockIdx.x >> 4;
    const int chunk = blockIdx.x & 15;
    const int p = threadIdx.x;
    const float* asn = dout + ASN_OFF;

    float ap = 0.f, ge = 0.f;
    const int n0 = chunk * 128;
    for (int i = 0; i < 128; ++i) {
        const size_t row = (size_t)b * N_ + (n0 + i);
        const float a0 = asn[row * 2];
        const float xv = x[row * P3_ + p];
        ap = fmaf(a0, xv, ap);
        ge += xv;
    }
    ws[WS_POS + (size_t)blockIdx.x * 128 + p] = ap;
    ws[WS_GE  + (size_t)blockIdx.x * 128 + p] = ge;
}

// Kernel 2b: deterministic chunk reduction -> means
__global__ void __launch_bounds__(128)
k_embed_red(const float* __restrict__ ws, float* __restrict__ out)
{
    const int b = blockIdx.x, p = threadIdx.x;
    float sp = 0.f, sg = 0.f;
    #pragma unroll
    for (int c = 0; c < 16; ++c) {
        sp += ws[WS_POS + ((size_t)(b * 16 + c)) * 128 + p];
        sg += ws[WS_GE  + ((size_t)(b * 16 + c)) * 128 + p];
    }
    const float scale = 1.f / (float)N_;
    out[POS_OFF + b * 128 + p] = sp * scale;
    out[GE_OFF  + b * 128 + p] = sg * scale;
}

// -------------------------------------------------------------------------
// Kernel 3: new_adj[b] = S^T A S via edge list (duplicates sum naturally),
// L1-normalize rows, per-graph diag penalty -> ws. 64 blocks x 256 threads.
// -------------------------------------------------------------------------
__global__ void __launch_bounds__(256)
k_pool(const long long* __restrict__ ei, const float* __restrict__ dout,
       float* __restrict__ ws)
{
    __shared__ float sred[8][4];
    const int b = blockIdx.x, tid = threadIdx.x;
    const float* asn = dout + ASN_OFF + (size_t)b * N_ * 2;

    float c00 = 0.f, c01 = 0.f, c10 = 0.f, c11 = 0.f;
    for (int e = tid; e < E_; e += 256) {
        const int s = (int)ei[e];
        const int d = (int)ei[E_ + e];
        const float s0 = asn[s * 2], s1 = asn[s * 2 + 1];
        const float d0 = asn[d * 2], d1 = asn[d * 2 + 1];
        c00 = fmaf(s0, d0, c00); c01 = fmaf(s0, d1, c01);
        c10 = fmaf(s1, d0, c10); c11 = fmaf(s1, d1, c11);
    }
    #pragma unroll
    for (int off = 16; off >= 1; off >>= 1) {
        c00 += __shfl_xor(c00, off, 32); c01 += __shfl_xor(c01, off, 32);
        c10 += __shfl_xor(c10, off, 32); c11 += __shfl_xor(c11, off, 32);
    }
    if ((tid & 31) == 0) {
        sred[tid >> 5][0] = c00; sred[tid >> 5][1] = c01;
        sred[tid >> 5][2] = c10; sred[tid >> 5][3] = c11;
    }
    __syncthreads();
    if (tid == 0) {
        float t00 = 0.f, t01 = 0.f, t10 = 0.f, t11 = 0.f;
        #pragma unroll
        for (int w = 0; w < 8; ++w) {
            t00 += sred[w][0]; t01 += sred[w][1];
            t10 += sred[w][2]; t11 += sred[w][3];
        }
        const float n00 = t00 / fmaxf(fabsf(t00) + fabsf(t01), 1e-5f);
        const float n11 = t11 / fmaxf(fabsf(t10) + fabsf(t11), 1e-5f);
        ws[WS_PEN + b] = (n00 - 1.f) * (n00 - 1.f) + (n11 - 1.f) * (n11 - 1.f);
    }
}

// Kernel 3b: final penalty scalar (deterministic single-thread sum of 64)
__global__ void k_pen(const float* __restrict__ ws, float* __restrict__ out)
{
    if (threadIdx.x == 0 && blockIdx.x == 0) {
        float s = 0.f;
        for (int b = 0; b < B_; ++b) s += ws[WS_PEN + b];
        out[PEN_OFF] = s * (1.f / (float)(B_ * 2));
    }
}

// -------------------------------------------------------------------------
// Kernel 4: out = fc2(relu(fc1(pos_embedding))). 64 blocks x 128 threads.
// -------------------------------------------------------------------------
__global__ void __launch_bounds__(128)
k_fc(const float* __restrict__ Wfc1, const float* __restrict__ bfc1,
     const float* __restrict__ Wfc2, const float* __restrict__ bfc2,
     float* __restrict__ out)
{
    __shared__ float sPos[128];
    __shared__ float sO[128];
    const int b = blockIdx.x, t = threadIdx.x;
    sPos[t] = out[POS_OFF + b * 128 + t];
    __syncthreads();
    float s = bfc1[t];
    const float* w1 = Wfc1 + (size_t)t * 128;
    #pragma unroll 8
    for (int k = 0; k < 128; ++k) s = fmaf(sPos[k], w1[k], s);
    sO[t] = fmaxf(s, 0.f);
    __syncthreads();
    float s2 = bfc2[t];
    const float* w2 = Wfc2 + (size_t)t * 128;
    #pragma unroll 8
    for (int k = 0; k < 128; ++k) s2 = fmaf(sO[k], w2[k], s2);
    out[OUT_OFF + b * 128 + t] = s2;
}

extern "C" void kernel_launch(void* const* d_in, const int* in_sizes, int n_in,
                              void* d_out, int out_size, void* d_ws, size_t ws_size,
                              hipStream_t stream) {
    const float*     emb  = (const float*)d_in[0];
    const float*     x    = (const float*)d_in[1];
    const long long* ei   = (const long long*)d_in[2];
    // d_in[3] = batch (unused: arange, one graph per batch slice)
    const float* Wc1  = (const float*)d_in[4];
    const float* bc1  = (const float*)d_in[5];
    const float* Wc2  = (const float*)d_in[6];
    const float* bc2  = (const float*)d_in[7];
    const float* Wfc1 = (const float*)d_in[8];
    const float* bfc1 = (const float*)d_in[9];
    const float* Wfc2 = (const float*)d_in[10];
    const float* bfc2 = (const float*)d_in[11];
    float* out = (float*)d_out;
    float* ws  = (float*)d_ws;

    k_assign   <<<(B_ * N_) / 128, 256, 0, stream>>>(emb, Wc1, bc1, Wc2, bc2, out);
    k_embed_part<<<B_ * 16, 128, 0, stream>>>(x, out, ws);
    k_embed_red <<<B_, 128, 0, stream>>>(ws, out);
    k_pool     <<<B_, 256, 0, stream>>>(ei, out, ws);
    k_pen      <<<1, 32, 0, stream>>>(ws, out);
    k_fc       <<<B_, 128, 0, stream>>>(Wfc1, bfc1, Wfc2, bfc2, out);
}
